// MultiHeadDilatedLocalAttention_31791347925504
// MI455X (gfx1250) — matrix operand order repaired and checked by reading, not compiled, not run on hardware
//
#include <hip/hip_runtime.h>

// Problem constants (match reference)
#define Bsz   2
#define Ssz   2048
#define DMsz  1024
#define Hsz   16
#define HDsz  64
#define NEGV  (-1e9f)

typedef __attribute__((ext_vector_type(16))) __bf16 v16bf;
typedef __attribute__((ext_vector_type(8)))  float  v8f;
typedef __attribute__((ext_vector_type(4)))  unsigned int v4u;
typedef __attribute__((ext_vector_type(8)))  unsigned int v8u;

union FragBF {
  v16bf bf;
  unsigned int u[8];
  uint4 u4[2];
};

__device__ __forceinline__ unsigned short f2bf(float f) {
  unsigned int x = __float_as_uint(f);
  x += 0x7FFFu + ((x >> 16) & 1u);   // round-to-nearest-even
  return (unsigned short)(x >> 16);
}
__device__ __forceinline__ unsigned int pack2(float lo, float hi) {
  return (unsigned int)f2bf(lo) | ((unsigned int)f2bf(hi) << 16);
}
__device__ __forceinline__ v8f wmma_bf16(const FragBF& a, const FragBF& b, v8f c) {
  return __builtin_amdgcn_wmma_f32_16x16x32_bf16(false, a.bf, false, b.bf,
                                                 (short)0, c, false, false);
}

// ---------------------------------------------------------------------------
// Setup: expand global_idx into byte flags g[S] and per-32-key-chunk flags
// ---------------------------------------------------------------------------
__global__ void init_flags(unsigned char* g, int* cf) {
  int i = blockIdx.x * blockDim.x + threadIdx.x;
  if (i < Ssz) g[i] = 0;
  if (i < Ssz / 32) cf[i] = 0;
}
__global__ void set_flags(const int* __restrict__ gidx, unsigned char* g, int* cf) {
  int t = threadIdx.x;
  if (t < 32) {
    int i = gidx[t];
    g[i] = 1;
    atomicOr(&cf[i >> 5], 1);
  }
}

// ---------------------------------------------------------------------------
// GEMM: out[m,n] = sum_k A[m,k] * W[n,k] + bias[n]
// Register-blocked: each wave computes a 16x64 output strip. Per k-step one
// A-fragment is reused across 4 B-fragments -> 4 WMMAs (16x16x32 bf16,
// f32 accumulate). A traffic amortized 4x; WMMA:VMEM issue ratio doubled.
// ---------------------------------------------------------------------------
template<bool A_BF16, bool OUT_BF16>
__global__ __launch_bounds__(256)
void gemm_wmma(const void* __restrict__ Ain, const float* __restrict__ W,
               const float* __restrict__ bias, void* __restrict__ out,
               int M, int N, int K) {
  const int lane  = threadIdx.x & 31;
  const int wave  = threadIdx.x >> 5;
  const int m0    = blockIdx.x * 16;
  const int n0    = (blockIdx.y * 8 + wave) * 64;   // 64-col strip per wave
  const int c     = lane & 15;
  const int hi    = lane >> 4;
  const int khalf = hi * 8;

  const float* wRow0 = W + (size_t)(n0 + c) * K;
  v8f acc0 = {0.f,0.f,0.f,0.f,0.f,0.f,0.f,0.f};
  v8f acc1 = acc0, acc2 = acc0, acc3 = acc0;

  for (int k0 = 0; k0 < K; k0 += 32) {
    FragBF a;
    if (A_BF16) {
      const unsigned short* ap =
          (const unsigned short*)Ain + (size_t)(m0 + c) * K + k0;
      a.u4[0] = *(const uint4*)(ap + khalf);
      a.u4[1] = *(const uint4*)(ap + 16 + khalf);
    } else {
      const float* ap = (const float*)Ain + (size_t)(m0 + c) * K + k0;
      float4 f0 = *(const float4*)(ap + khalf);
      float4 f1 = *(const float4*)(ap + khalf + 4);
      float4 f2 = *(const float4*)(ap + 16 + khalf);
      float4 f3 = *(const float4*)(ap + 16 + khalf + 4);
      a.u[0] = pack2(f0.x, f0.y); a.u[1] = pack2(f0.z, f0.w);
      a.u[2] = pack2(f1.x, f1.y); a.u[3] = pack2(f1.z, f1.w);
      a.u[4] = pack2(f2.x, f2.y); a.u[5] = pack2(f2.z, f2.w);
      a.u[6] = pack2(f3.x, f3.y); a.u[7] = pack2(f3.z, f3.w);
    }
    #pragma unroll
    for (int t = 0; t < 4; t++) {
      FragBF b;
      const float* wp = wRow0 + (size_t)(t * 16) * K + k0 + khalf;
      float4 g0 = *(const float4*)(wp);
      float4 g1 = *(const float4*)(wp + 4);
      float4 g2 = *(const float4*)(wp + 16);
      float4 g3 = *(const float4*)(wp + 20);
      b.u[0] = pack2(g0.x, g0.y); b.u[1] = pack2(g0.z, g0.w);
      b.u[2] = pack2(g1.x, g1.y); b.u[3] = pack2(g1.z, g1.w);
      b.u[4] = pack2(g2.x, g2.y); b.u[5] = pack2(g2.z, g2.w);
      b.u[6] = pack2(g3.x, g3.y); b.u[7] = pack2(g3.z, g3.w);
      if      (t == 0) acc0 = wmma_bf16(a, b, acc0);
      else if (t == 1) acc1 = wmma_bf16(a, b, acc1);
      else if (t == 2) acc2 = wmma_bf16(a, b, acc2);
      else             acc3 = wmma_bf16(a, b, acc3);
    }
  }

  #pragma unroll
  for (int t = 0; t < 4; t++) {
    const int n = n0 + t * 16 + c;
    const float bb = bias[n];
    #pragma unroll
    for (int v = 0; v < 8; v++) {
      const int r = m0 + v + hi * 8;
      const size_t off = (size_t)r * N + n;
      const v8f& acc = (t == 0) ? acc0 : (t == 1) ? acc1 : (t == 2) ? acc2 : acc3;
      const float val = acc[v] + bb;
      if (OUT_BF16) ((unsigned short*)out)[off] = f2bf(val);
      else          ((float*)out)[off] = val;
    }
  }
}

// ---------------------------------------------------------------------------
// Flash-style dilated local + global attention.
// One wave per (b, h, 16-query tile). 32 keys per step; fully-masked key
// chunks are skipped (wave-uniform branch). V tile staged into LDS by the
// Tensor Data Mover (TDM, TENSORcnt-tracked); online softmax; PV via WMMA.
// ---------------------------------------------------------------------------
__global__ __launch_bounds__(32)
void attn_wmma(const unsigned short* __restrict__ Qh,
               const unsigned short* __restrict__ Kh,
               const unsigned short* __restrict__ Vh,
               const unsigned char*  __restrict__ g,
               const int*            __restrict__ cf,
               unsigned short*       __restrict__ AO) {
  const int qt = blockIdx.x, h = blockIdx.y, b = blockIdx.z;
  const int lane  = threadIdx.x;
  const int c     = lane & 15;
  const int hi    = lane >> 4;
  const int khalf = hi * 8;
  const int q0    = qt * 16;

  __shared__ unsigned char  gq[16];
  __shared__ unsigned short sP[16 * 32];       // probabilities, row-major
  __shared__ unsigned short sV[32 * HDsz];     // V tile, row-major (TDM dest)

  unsigned char myg = (lane < 16) ? g[q0 + lane] : (unsigned char)0;
  if (lane < 16) gq[lane] = myg;
  const bool qHasGlobal = (__ballot(myg != 0) != 0ull);
  __syncthreads();

  // Q fragments (16 rows x 64 d, two 16x32 A-fragments)
  FragBF aq0, aq1;
  {
    const unsigned short* qp =
        Qh + ((size_t)(b * Ssz + q0 + c)) * DMsz + h * HDsz;
    aq0.u4[0] = *(const uint4*)(qp + khalf);
    aq0.u4[1] = *(const uint4*)(qp + 16 + khalf);
    aq1.u4[0] = *(const uint4*)(qp + 32 + khalf);
    aq1.u4[1] = *(const uint4*)(qp + 48 + khalf);
  }

  // TDM descriptor invariants (D# group1): 2D tile, 2-byte elements,
  // tile 64 x 32, tensor_dim0 = 64, row stride = DMsz elements.
  v8u dg1;
  dg1[0] = 1u << 16;                 // data_size=1 (2 bytes), no multicast
  dg1[1] = 64u << 16;                // tensor_dim0[15:0] = 64
  dg1[2] = 32u << 16;                // tensor_dim0 hi = 0 ; tensor_dim1[15:0] = 32
  dg1[3] = 64u << 16;                // tensor_dim1 hi = 0 ; tile_dim0 = 64
  dg1[4] = 32u;                      // tile_dim1 = 32 ; tile_dim2 = 0
  dg1[5] = (unsigned)DMsz;           // tensor_dim0_stride[31:0] = 1024
  dg1[6] = 0u;                       // stride hi ; tensor_dim1_stride lo
  dg1[7] = 0u;                       // tensor_dim1_stride hi
  const unsigned lds_off = (unsigned)(unsigned long long)(void*)sV;

  float m[8], l[8];
  #pragma unroll
  for (int v = 0; v < 8; v++) { m[v] = -1e30f; l[v] = 0.0f; }
  v8f o0 = {0.f,0.f,0.f,0.f,0.f,0.f,0.f,0.f};
  v8f o1 = o0, o2 = o0, o3 = o0;

  for (int j0 = 0; j0 < Ssz; j0 += 32) {
    // wave-uniform skip: chunk overlaps the dilated window or has global cols
    const bool wov = (j0 + 31 >= q0 - 512) && (j0 <= q0 + 527);
    if (!qHasGlobal && !wov && (cf[j0 >> 5] == 0)) continue;

    // --- kick off V-tile DMA (global -> LDS) via Tensor Data Mover ---
    {
      // WAR: previous step's LDS reads of sV must retire before TDM rewrites
      asm volatile("s_wait_dscnt 0x0" ::: "memory");
      const unsigned long long ga = (unsigned long long)(const void*)
          (Vh + ((size_t)(b * Ssz + j0)) * DMsz + h * HDsz);
      v4u dg0;
      dg0[0] = 1u;                                   // count=1, user descriptor
      dg0[1] = lds_off;                              // lds_addr
      dg0[2] = (unsigned)ga;                         // global_addr[31:0]
      dg0[3] = (unsigned)((ga >> 32) & 0x1FFFFFFu)   // global_addr[56:32]
             | (2u << 30);                           // type = 2 ("image")
      asm volatile("tensor_load_to_lds %0, %1" :: "s"(dg0), "s"(dg1) : "memory");
    }

    // --- scores S = Q K^T (two 16-key tiles) ---
    const unsigned short* kp0 =
        Kh + ((size_t)(b * Ssz + j0 + c)) * DMsz + h * HDsz;
    const unsigned short* kp1 = kp0 + (size_t)16 * DMsz;
    FragBF bk00, bk01, bk10, bk11;
    bk00.u4[0] = *(const uint4*)(kp0 + khalf);
    bk00.u4[1] = *(const uint4*)(kp0 + 16 + khalf);
    bk01.u4[0] = *(const uint4*)(kp0 + 32 + khalf);
    bk01.u4[1] = *(const uint4*)(kp0 + 48 + khalf);
    bk10.u4[0] = *(const uint4*)(kp1 + khalf);
    bk10.u4[1] = *(const uint4*)(kp1 + 16 + khalf);
    bk11.u4[0] = *(const uint4*)(kp1 + 32 + khalf);
    bk11.u4[1] = *(const uint4*)(kp1 + 48 + khalf);
    // prefetch next key chunk (speculative; lowers to global_prefetch_b8)
    __builtin_prefetch((const void*)(kp0 + (size_t)32 * DMsz), 0, 0);

    v8f s0 = {0.f,0.f,0.f,0.f,0.f,0.f,0.f,0.f};
    v8f s1 = s0;
    s0 = wmma_bf16(aq0, bk00, s0);
    s0 = wmma_bf16(aq1, bk01, s0);
    s1 = wmma_bf16(aq0, bk10, s1);
    s1 = wmma_bf16(aq1, bk11, s1);

    // --- mask + scale + online softmax (C layout: row = v+8*hi, col = c) ---
    const unsigned char gca = g[j0 + c];
    const unsigned char gcb = g[j0 + 16 + c];
    #pragma unroll
    for (int v = 0; v < 8; v++) {
      const int r = v + hi * 8;
      const int i = q0 + r;
      const bool rowg = (gq[r] != 0);
      const int da = (j0 + c) - i;
      const int db = (j0 + 16 + c) - i;
      const bool ma = rowg || (gca != 0) ||
                      ((da >= -512) && (da <= 512) && ((da & 1) == 0));
      const bool mb = rowg || (gcb != 0) ||
                      ((db >= -512) && (db <= 512) && ((db & 1) == 0));
      const float sa = ma ? s0[v] * 0.125f : NEGV;  // 1/sqrt(64)
      const float sb = mb ? s1[v] * 0.125f : NEGV;

      float mx = fmaxf(sa, sb);
      #pragma unroll
      for (int off2 = 1; off2 < 16; off2 <<= 1)
        mx = fmaxf(mx, __shfl_xor(mx, off2, 16));
      const float mnew  = fmaxf(m[v], mx);
      const float alpha = __expf(m[v] - mnew);
      const float ea = __expf(sa - mnew);
      const float eb = __expf(sb - mnew);
      float rs = ea + eb;
      #pragma unroll
      for (int off2 = 1; off2 < 16; off2 <<= 1)
        rs += __shfl_xor(rs, off2, 16);
      l[v] = l[v] * alpha + rs;
      m[v] = mnew;
      o0[v] *= alpha; o1[v] *= alpha; o2[v] *= alpha; o3[v] *= alpha;
      sP[r * 32 + c]      = f2bf(ea);
      sP[r * 32 + c + 16] = f2bf(eb);
    }
    __syncthreads();

    // --- P fragment in A layout (row = lane%16, K striped) ---
    FragBF ap;
    {
      const uint4* pr = (const uint4*)&sP[c * 32];
      ap.u4[0] = pr[2 * hi];       // K = 16*hi .. 16*hi+7
      ap.u4[1] = pr[2 * hi + 1];   // K = 16*hi+8 ..
    }

    // --- wait for TDM, then O += P x V : four 16-d output tiles ---
    __builtin_amdgcn_s_wait_tensorcnt(0);
    #pragma unroll
    for (int dt = 0; dt < 4; dt++) {
      FragBF vb;
      const int d   = dt * 16 + c;
      const int kkb = hi * 16;
      #pragma unroll
      for (int v = 0; v < 8; v++) {
        vb.u[v] = (unsigned int)sV[(kkb + 2 * v) * HDsz + d]
                | ((unsigned int)sV[(kkb + 2 * v + 1) * HDsz + d] << 16);
      }
      if      (dt == 0) o0 = wmma_bf16(ap, vb, o0);
      else if (dt == 1) o1 = wmma_bf16(ap, vb, o1);
      else if (dt == 2) o2 = wmma_bf16(ap, vb, o2);
      else              o3 = wmma_bf16(ap, vb, o3);
    }
    __syncthreads();
  }

  // normalize and store bf16 [B,S,DM] (head-interleaved layout)
  #pragma unroll
  for (int v = 0; v < 8; v++) {
    const float inv = 1.0f / l[v];
    const int r = q0 + v + hi * 8;
    unsigned short* op = AO + ((size_t)(b * Ssz + r)) * DMsz + h * HDsz;
    op[c]      = f2bf(o0[v] * inv);
    op[16 + c] = f2bf(o1[v] * inv);
    op[32 + c] = f2bf(o2[v] * inv);
    op[48 + c] = f2bf(o3[v] * inv);
  }
}

// ---------------------------------------------------------------------------
// Launch
// ---------------------------------------------------------------------------
extern "C" void kernel_launch(void* const* d_in, const int* in_sizes, int n_in,
                              void* d_out, int out_size, void* d_ws, size_t ws_size,
                              hipStream_t stream) {
  const float* q   = (const float*)d_in[0];
  const float* k   = (const float*)d_in[1];
  const float* v   = (const float*)d_in[2];
  const float* Wq  = (const float*)d_in[3];
  const float* Wk  = (const float*)d_in[4];
  const float* Wv  = (const float*)d_in[5];
  const float* Wo  = (const float*)d_in[6];
  const float* bq  = (const float*)d_in[7];
  const float* bk  = (const float*)d_in[8];
  const float* bv  = (const float*)d_in[9];
  const float* bo  = (const float*)d_in[10];
  const int*   gidx = (const int*)d_in[11];
  float* out = (float*)d_out;

  const size_t NE = (size_t)Bsz * Ssz * DMsz;   // 4,194,304 elements
  char* ws = (char*)d_ws;
  unsigned short* Qh = (unsigned short*)(ws);
  unsigned short* Kh = (unsigned short*)(ws + NE * 2);
  unsigned short* Vh = (unsigned short*)(ws + NE * 4);
  unsigned short* AO = (unsigned short*)(ws + NE * 6);
  unsigned char*  gf = (unsigned char*)(ws + NE * 8);
  int*            cf = (int*)(ws + NE * 8 + 2048);

  init_flags<<<dim3((Ssz + 255) / 256), dim3(256), 0, stream>>>(gf, cf);
  set_flags<<<dim3(1), dim3(32), 0, stream>>>(gidx, gf, cf);

  // 8 waves per block, each wave owns a 16x64 strip: grid (M/16, N/512)
  const dim3 gg(Bsz * Ssz / 16, DMsz / 512);    // (256, 2)
  const dim3 gb(256);
  gemm_wmma<false, true><<<gg, gb, 0, stream>>>(q, Wq, bq, Qh, Bsz * Ssz, DMsz, DMsz);
  gemm_wmma<false, true><<<gg, gb, 0, stream>>>(k, Wk, bk, Kh, Bsz * Ssz, DMsz, DMsz);
  gemm_wmma<false, true><<<gg, gb, 0, stream>>>(v, Wv, bv, Vh, Bsz * Ssz, DMsz, DMsz);

  attn_wmma<<<dim3(Ssz / 16, Hsz, Bsz), dim3(32), 0, stream>>>(Qh, Kh, Vh, gf, cf, AO);

  gemm_wmma<true, false><<<gg, gb, 0, stream>>>(AO, Wo, bo, out, Bsz * Ssz, DMsz, DMsz);
}
